// SSR2Block_44032004718755
// MI455X (gfx1250) — matrix operand order repaired and checked by reading, not compiled, not run on hardware
//
#include <hip/hip_runtime.h>
#include <hip/hip_bf16.h>
#include <math.h>

#define B_DIM  1024
#define NS_DIM 32
#define D_DIM  1024
#define H_DIM  4096
#define LN_EPS 1e-5f

typedef __attribute__((ext_vector_type(16))) __bf16 v16bf;
typedef __attribute__((ext_vector_type(8)))  __bf16 v8bf;
typedef __attribute__((ext_vector_type(8)))  float  v8f;
typedef __attribute__((ext_vector_type(4)))  int    v4i;

// Pointer types matching the gfx1250 async-LDS builtin signature:
//   void __builtin_amdgcn_global_load_async_to_lds_b128(v4i AS1*, v4i AS3*, Ii, Ii)
typedef __attribute__((address_space(1))) v4i* gptr_v4i;
typedef __attribute__((address_space(3))) v4i* lptr_v4i;

// CDNA5 async global->LDS path (guarded: falls back to sync copy if the
// toolchain doesn't declare the builtins).
#if defined(__has_builtin)
#if __has_builtin(__builtin_amdgcn_global_load_async_to_lds_b128) && \
    __has_builtin(__builtin_amdgcn_s_wait_asynccnt)
#define HAVE_ASYNC_LDS 1
#endif
#endif

// ---------------------------------------------------------------- utilities
__device__ __forceinline__ unsigned short f32_to_bf16(float f) {
  unsigned int u = __float_as_uint(f);
  u += 0x7FFFu + ((u >> 16) & 1u);   // round-to-nearest-even
  return (unsigned short)(u >> 16);
}

__device__ __forceinline__ float gelu_exact(float v) {
  return 0.5f * v * (1.0f + erff(v * 0.70710678118654752f));
}

// A fragment (16x32 bf16, row-major A[M,K]): lane half h, row = lane&15.
// v16bf elements 0..7  = K(k0+16h .. k0+16h+7)
// v16bf elements 8..15 = K(k0+16h+8 .. k0+16h+15)
__device__ __forceinline__ v16bf load_a_frag(const unsigned short* __restrict__ A,
                                             int row, int K, int k0, int h) {
  union { v16bf v; struct { v8bf lo, hi; } s; } u;
  const unsigned short* base = A + (size_t)row * K + k0 + 16 * h;
  u.s.lo = *reinterpret_cast<const v8bf*>(base);
  u.s.hi = *reinterpret_cast<const v8bf*>(base + 8);
  return u.v;
}

// B fragment (32x16 bf16) from pre-transposed weights Bt[N,K]:
// lane half h, col n = lane&15; elements 0..15 = K(k0+16h .. k0+16h+15)
__device__ __forceinline__ v16bf load_b_frag(const unsigned short* __restrict__ Bt,
                                             int n, int K, int k0, int h) {
  return *reinterpret_cast<const v16bf*>(Bt + (size_t)n * K + k0 + 16 * h);
}

#define WMMA_BF16(a, b, c) \
  __builtin_amdgcn_wmma_f32_16x16x32_bf16(false, (a), false, (b), (short)0, (c), false, false)

// ------------------------------------------------------- weight preparation
// dst[n*K + k] = bf16(src[k*N + n])   (transpose + convert, LDS tiled)
__global__ __launch_bounds__(256) void transpose_to_bf16_kernel(
    const float* __restrict__ src, unsigned short* __restrict__ dst, int K, int N) {
  __shared__ unsigned short tile[32][33];
  int tx = threadIdx.x & 31, ty = threadIdx.x >> 5;
  int kb = blockIdx.y * 32, nb = blockIdx.x * 32;
  #pragma unroll
  for (int i = ty; i < 32; i += 8) {
    int k = kb + i, n = nb + tx;
    tile[i][tx] = (k < K && n < N) ? f32_to_bf16(src[(size_t)k * N + n]) : (unsigned short)0;
  }
  __syncthreads();
  #pragma unroll
  for (int i = ty; i < 32; i += 8) {
    int n = nb + i, k = kb + tx;
    if (n < N && k < K) dst[(size_t)n * K + k] = tile[tx][i];
  }
}

__global__ __launch_bounds__(256) void convert_bf16_kernel(
    const float* __restrict__ src, unsigned short* __restrict__ dst, size_t n) {
  size_t i = (size_t)blockIdx.x * 256 + threadIdx.x;
  size_t stride = (size_t)gridDim.x * 256;
  for (; i < n; i += stride) dst[i] = f32_to_bf16(src[i]);
}

// ----------------------------------------------------- routing: ctx / softmax
// ctx[b, 0:D] = x[b], ctx[b, D:2D] = mean_s S[b,s,:]
__global__ __launch_bounds__(256) void build_ctx_kernel(
    const float* __restrict__ x, const float* __restrict__ S,
    unsigned short* __restrict__ ctx) {
  int idx = blockIdx.x * 256 + threadIdx.x;           // over B*D
  int b = idx >> 10, d = idx & (D_DIM - 1);
  float s = 0.f;
  #pragma unroll 4
  for (int si = 0; si < NS_DIM; ++si)
    s += S[((size_t)b * NS_DIM + si) * D_DIM + d];
  ctx[(size_t)b * 2 * D_DIM + d]         = f32_to_bf16(x[idx]);
  ctx[(size_t)b * 2 * D_DIM + D_DIM + d] = f32_to_bf16(s * (1.0f / NS_DIM));
}

__global__ __launch_bounds__(256) void softmax_kernel(
    const float* __restrict__ logits, float* __restrict__ alpha, int rows) {
  int wid = threadIdx.x >> 5, lane = threadIdx.x & 31;
  int row = blockIdx.x * 8 + wid;
  if (row >= rows) return;
  float v = logits[row * NS_DIM + lane];
  float m = v;
  #pragma unroll
  for (int off = 16; off > 0; off >>= 1) m = fmaxf(m, __shfl_xor(m, off, 32));
  float e = expf(v - m);
  float s = e;
  #pragma unroll
  for (int off = 16; off > 0; off >>= 1) s += __shfl_xor(s, off, 32);
  alpha[row * NS_DIM + lane] = e / s;
}

// ---------------------------------------------------------- generic WMMA GEMM
// C[M,N] = A[M,K](bf16) * Bt[N,K](bf16)^T + bias, epilogue EP.
// block = 256 thr = 8 waves (4 M x 2 N); block tile 64x64; wave tile 16x32.
enum { EP_STORE_F32 = 0, EP_GELU_BF16 = 2, EP_ADD_F32 = 3 };

template <int EP>
__global__ __launch_bounds__(256) void gemm_wmma_bf16(
    const unsigned short* __restrict__ A, const unsigned short* __restrict__ Bt,
    const float* __restrict__ bias, const float* __restrict__ res,
    void* __restrict__ out, int M, int N, int K) {
  int tid = threadIdx.x, wid = tid >> 5, lane = tid & 31;
  int h = lane >> 4, l16 = lane & 15;
  int wm = wid & 3, wn = wid >> 2;
  int rowBase = blockIdx.y * 64 + wm * 16;
  int colBase = blockIdx.x * 64 + wn * 32;
  if (colBase >= N) return;                 // wave-uniform guard (EXEC stays full)
  int arow = rowBase + l16;
  int n0 = colBase + l16, n1 = colBase + 16 + l16;

  v8f acc0 = {}, acc1 = {};
  for (int k0 = 0; k0 < K; k0 += 32) {
    v16bf a  = load_a_frag(A, arow, K, k0, h);
    v16bf b0 = load_b_frag(Bt, n0, K, k0, h);
    v16bf b1 = load_b_frag(Bt, n1, K, k0, h);
    __builtin_prefetch(A + (size_t)arow * K + k0 + 32, 0, 3);
    __builtin_prefetch(Bt + (size_t)n0 * K + k0 + 32, 0, 3);
    acc0 = WMMA_BF16(a, b0, acc0);
    acc1 = WMMA_BF16(a, b1, acc1);
  }

  float bias0 = bias ? bias[n0] : 0.f;
  float bias1 = bias ? bias[n1] : 0.f;
  #pragma unroll
  for (int r = 0; r < 8; ++r) {
    int grow = rowBase + 8 * h + r;
    size_t i0 = (size_t)grow * N + n0;
    size_t i1 = (size_t)grow * N + n1;
    float v0 = acc0[r] + bias0;
    float v1 = acc1[r] + bias1;
    if constexpr (EP == EP_STORE_F32) {
      ((float*)out)[i0] = v0;
      ((float*)out)[i1] = v1;
    } else if constexpr (EP == EP_GELU_BF16) {
      ((unsigned short*)out)[i0] = f32_to_bf16(gelu_exact(v0));
      ((unsigned short*)out)[i1] = f32_to_bf16(gelu_exact(v1));
    } else { // EP_ADD_F32
      ((float*)out)[i0] = v0 + res[i0];
      ((float*)out)[i1] = v1 + res[i1];
    }
  }
}

// ------------------------------------------- fused GRU GEMM (gh = S @ w_hh)
// 3 gate tiles per wave share one A tile; GRU nonlinearity fused in epilogue.
// B tiles (identical for all 8 waves) are staged in LDS once per block per
// K-chunk, via CDNA5 async global->LDS copies when available.
// block = 8 waves stacked in M (128 rows), 16 cols of D. grid=(D/16, M/128)
#define GRU_KC 512   // K-chunk staged in LDS: 3 gates * 16 cols * 512 * 2B = 48KB

__global__ __launch_bounds__(256) void gru_gemm_wmma(
    const unsigned short* __restrict__ Sb,   // [B*NS, D] bf16
    const unsigned short* __restrict__ Bt,   // [3D, D]  bf16 (w_hh transposed)
    const float* __restrict__ alpha,         // [B*NS]
    const float* __restrict__ xW,            // [B, 3D]  (x @ w_ih, no bias)
    const float* __restrict__ b_ih, const float* __restrict__ b_hh,
    const float* __restrict__ Sold,          // [B*NS, D] f32
    float* __restrict__ Snew) {
  __shared__ __align__(32) unsigned short ldsB[3 * 16 * GRU_KC];
  int tid = threadIdx.x, wid = tid >> 5, lane = tid & 31;
  int h = lane >> 4, l16 = lane & 15;
  int rowBase = blockIdx.y * 128 + wid * 16;
  int col0 = blockIdx.x * 16;
  int j = col0 + l16;
  int arow = rowBase + l16;
  const int K = D_DIM, D = D_DIM;

  v8f ar = {}, az = {}, an = {};
  for (int kc = 0; kc < K; kc += GRU_KC) {
    // ---- cooperative stage of B chunk: 3 gates x 16 cols x GRU_KC ----
    {
      const int CPR = GRU_KC / 8;               // 16B chunks per row (=64)
      const int chunks = 3 * 16 * CPR;          // total 16B chunks (=3072)
      #pragma unroll 1
      for (int ci = tid; ci < chunks; ci += 256) {
        int r = ci / CPR;                       // row index: gate*16 + col
        int w = ci - r * CPR;                   // chunk within row
        int g = r >> 4, c = r & 15;
        const unsigned short* gsrc =
            Bt + (size_t)(g * D + col0 + c) * K + kc + w * 8;
        unsigned short* ldst = &ldsB[r * GRU_KC + w * 8];
#if defined(HAVE_ASYNC_LDS)
        __builtin_amdgcn_global_load_async_to_lds_b128(
            (gptr_v4i)gsrc, (lptr_v4i)ldst, 0, 0);
#else
        *reinterpret_cast<v8bf*>(ldst) = *reinterpret_cast<const v8bf*>(gsrc);
#endif
      }
#if defined(HAVE_ASYNC_LDS)
      __builtin_amdgcn_s_wait_asynccnt(0);
#endif
      __syncthreads();
    }
    // ---- compute over the staged chunk (unroll capped: avoid VGPR spills) ----
    #pragma unroll 4
    for (int k0 = 0; k0 < GRU_KC; k0 += 32) {
      v16bf a  = load_a_frag(Sb, arow, K, kc + k0, h);
      v16bf br = *reinterpret_cast<const v16bf*>(&ldsB[(0 * 16 + l16) * GRU_KC + k0 + 16 * h]);
      v16bf bz = *reinterpret_cast<const v16bf*>(&ldsB[(1 * 16 + l16) * GRU_KC + k0 + 16 * h]);
      v16bf bn = *reinterpret_cast<const v16bf*>(&ldsB[(2 * 16 + l16) * GRU_KC + k0 + 16 * h]);
      __builtin_prefetch(Sb + (size_t)arow * K + kc + k0 + 32, 0, 3);
      ar = WMMA_BF16(a, br, ar);
      az = WMMA_BF16(a, bz, az);
      an = WMMA_BF16(a, bn, an);
    }
    __syncthreads();
  }

  float bir = b_ih[j], biz = b_ih[D + j], bin_ = b_ih[2 * D + j];
  float bhr = b_hh[j], bhz = b_hh[D + j], bhn  = b_hh[2 * D + j];
  #pragma unroll
  for (int r = 0; r < 8; ++r) {
    int m = rowBase + 8 * h + r;            // m = b*NS + s
    int b = m >> 5;
    float al  = alpha[m];
    float gir = fmaf(al, xW[(size_t)b * 3 * D + j],         bir);
    float giz = fmaf(al, xW[(size_t)b * 3 * D + D + j],     biz);
    float gin = fmaf(al, xW[(size_t)b * 3 * D + 2 * D + j], bin_);
    float rr = 1.0f / (1.0f + expf(-(gir + ar[r] + bhr)));
    float zz = 1.0f / (1.0f + expf(-(giz + az[r] + bhz)));
    float nn = tanhf(gin + rr * (an[r] + bhn));
    size_t idx = (size_t)m * D + j;
    Snew[idx] = (1.0f - zz) * nn + zz * Sold[idx];
  }
}

// ----------------------------------------------- weighted readout over slots
__global__ __launch_bounds__(256) void readout_kernel(
    const float* __restrict__ alpha, const float* __restrict__ Snew,
    unsigned short* __restrict__ sc_bf16) {
  int idx = blockIdx.x * 256 + threadIdx.x;           // over B*D
  int b = idx >> 10, d = idx & (D_DIM - 1);
  float s = 0.f;
  #pragma unroll 4
  for (int si = 0; si < NS_DIM; ++si)
    s = fmaf(alpha[b * NS_DIM + si], Snew[((size_t)b * NS_DIM + si) * D_DIM + d], s);
  sc_bf16[idx] = f32_to_bf16(s);
}

// ------------------------------------------------------------------ LayerNorm
// one 256-thread block per row (D=1024, 4 elems/thread)
__global__ __launch_bounds__(256) void layernorm_kernel(
    const float* __restrict__ in, const float* __restrict__ gamma,
    const float* __restrict__ beta, float* __restrict__ outf,
    unsigned short* __restrict__ outb) {
  __shared__ float sbuf[256];
  int tid = threadIdx.x;
  const float* x = in + (size_t)blockIdx.x * D_DIM;
  float vals[4];
  float s = 0.f;
  #pragma unroll
  for (int i = 0; i < 4; ++i) { vals[i] = x[tid + 256 * i]; s += vals[i]; }
  sbuf[tid] = s; __syncthreads();
  for (int off = 128; off > 0; off >>= 1) {
    if (tid < off) sbuf[tid] += sbuf[tid + off];
    __syncthreads();
  }
  float mu = sbuf[0] * (1.0f / D_DIM); __syncthreads();
  float vs = 0.f;
  #pragma unroll
  for (int i = 0; i < 4; ++i) { float d = vals[i] - mu; vs += d * d; }
  sbuf[tid] = vs; __syncthreads();
  for (int off = 128; off > 0; off >>= 1) {
    if (tid < off) sbuf[tid] += sbuf[tid + off];
    __syncthreads();
  }
  float inv = rsqrtf(sbuf[0] * (1.0f / D_DIM) + LN_EPS);
  #pragma unroll
  for (int i = 0; i < 4; ++i) {
    int c = tid + 256 * i;
    float o = (vals[i] - mu) * inv * gamma[c] + beta[c];
    if (outf) outf[(size_t)blockIdx.x * D_DIM + c] = o;
    else      outb[(size_t)blockIdx.x * D_DIM + c] = f32_to_bf16(o);
  }
}

// ===================================================================== launch
extern "C" void kernel_launch(void* const* d_in, const int* in_sizes, int n_in,
                              void* d_out, int out_size, void* d_ws, size_t ws_size,
                              hipStream_t stream) {
  (void)in_sizes; (void)n_in; (void)out_size; (void)ws_size;
  const float* x      = (const float*)d_in[0];
  const float* S      = (const float*)d_in[1];
  const float* w_r1   = (const float*)d_in[2];
  const float* b_r1   = (const float*)d_in[3];
  const float* w_r2   = (const float*)d_in[4];
  const float* b_r2   = (const float*)d_in[5];
  const float* w_ih   = (const float*)d_in[6];
  const float* w_hh   = (const float*)d_in[7];
  const float* b_ih   = (const float*)d_in[8];
  const float* b_hh   = (const float*)d_in[9];
  const float* w_proj = (const float*)d_in[10];
  const float* b_proj = (const float*)d_in[11];
  const float* ln1_g  = (const float*)d_in[12];
  const float* ln1_b  = (const float*)d_in[13];
  const float* w_f1   = (const float*)d_in[14];
  const float* b_f1   = (const float*)d_in[15];
  const float* w_f2   = (const float*)d_in[16];
  const float* b_f2   = (const float*)d_in[17];
  const float* ln2_g  = (const float*)d_in[18];
  const float* ln2_b  = (const float*)d_in[19];

  float* out_x     = (float*)d_out;
  float* out_S     = out_x + (size_t)B_DIM * D_DIM;
  float* out_alpha = out_S + (size_t)B_DIM * NS_DIM * D_DIM;

  // ---- workspace carve-out
  char* ws = (char*)d_ws;
  size_t off = 0;
  auto alloc = [&](size_t bytes) -> void* {
    void* p = ws + off;
    off = (off + bytes + 255) & ~(size_t)255;
    return p;
  };
  unsigned short* w_r1t   = (unsigned short*)alloc((size_t)D_DIM * 2 * D_DIM * 2);
  unsigned short* w_r2t   = (unsigned short*)alloc((size_t)NS_DIM * D_DIM * 2);
  unsigned short* w_iht   = (unsigned short*)alloc((size_t)3 * D_DIM * D_DIM * 2);
  unsigned short* w_hht   = (unsigned short*)alloc((size_t)3 * D_DIM * D_DIM * 2);
  unsigned short* w_projt = (unsigned short*)alloc((size_t)D_DIM * D_DIM * 2);
  unsigned short* w_f1t   = (unsigned short*)alloc((size_t)H_DIM * D_DIM * 2);
  unsigned short* w_f2t   = (unsigned short*)alloc((size_t)D_DIM * H_DIM * 2);
  unsigned short* x_bf    = (unsigned short*)alloc((size_t)B_DIM * D_DIM * 2);
  unsigned short* S_bf    = (unsigned short*)alloc((size_t)B_DIM * NS_DIM * D_DIM * 2);
  unsigned short* ctx_bf  = (unsigned short*)alloc((size_t)B_DIM * 2 * D_DIM * 2);
  unsigned short* hr_bf   = (unsigned short*)alloc((size_t)B_DIM * D_DIM * 2);
  float*          rlog    = (float*)alloc((size_t)B_DIM * NS_DIM * 4);
  float*          xW      = (float*)alloc((size_t)B_DIM * 3 * D_DIM * 4);
  unsigned short* sc_bf   = (unsigned short*)alloc((size_t)B_DIM * D_DIM * 2);
  float*          x_mid   = (float*)alloc((size_t)B_DIM * D_DIM * 4);
  unsigned short* ln1_bf  = (unsigned short*)alloc((size_t)B_DIM * D_DIM * 2);
  unsigned short* ff1_bf  = (unsigned short*)alloc((size_t)B_DIM * H_DIM * 2);
  float*          tbuf    = (float*)alloc((size_t)B_DIM * D_DIM * 4);

  // ---- 1. weights: transpose + bf16  (Bt[n*K+k] = w[k*N+n])
  auto xpose = [&](const float* src, unsigned short* dst, int K, int N) {
    transpose_to_bf16_kernel<<<dim3((N + 31) / 32, (K + 31) / 32), 256, 0, stream>>>(src, dst, K, N);
  };
  xpose(w_r1,   w_r1t,   2 * D_DIM, D_DIM);
  xpose(w_r2,   w_r2t,   D_DIM,     NS_DIM);
  xpose(w_ih,   w_iht,   D_DIM,     3 * D_DIM);
  xpose(w_hh,   w_hht,   D_DIM,     3 * D_DIM);
  xpose(w_proj, w_projt, D_DIM,     D_DIM);
  xpose(w_f1,   w_f1t,   D_DIM,     H_DIM);
  xpose(w_f2,   w_f2t,   H_DIM,     D_DIM);

  // ---- 2. activations to bf16
  {
    size_t n1 = (size_t)B_DIM * D_DIM;
    convert_bf16_kernel<<<(int)((n1 + 255) / 256), 256, 0, stream>>>(x, x_bf, n1);
    size_t n2 = (size_t)B_DIM * NS_DIM * D_DIM;
    convert_bf16_kernel<<<(int)((n2 + 255) / 256), 256, 0, stream>>>(S, S_bf, n2);
  }

  // ---- 3. routing MLP + softmax
  build_ctx_kernel<<<(B_DIM * D_DIM) / 256, 256, 0, stream>>>(x, S, ctx_bf);
  gemm_wmma_bf16<EP_GELU_BF16><<<dim3(D_DIM / 64, B_DIM / 64), 256, 0, stream>>>(
      ctx_bf, w_r1t, b_r1, nullptr, hr_bf, B_DIM, D_DIM, 2 * D_DIM);
  gemm_wmma_bf16<EP_STORE_F32><<<dim3(1, B_DIM / 64), 256, 0, stream>>>(
      hr_bf, w_r2t, b_r2, nullptr, rlog, B_DIM, NS_DIM, D_DIM);
  softmax_kernel<<<B_DIM / 8, 256, 0, stream>>>(rlog, out_alpha, B_DIM);

  // ---- 4. GRU: xW = x @ w_ih (rank-1 trick), then fused gh-GEMM + gates
  gemm_wmma_bf16<EP_STORE_F32><<<dim3(3 * D_DIM / 64, B_DIM / 64), 256, 0, stream>>>(
      x_bf, w_iht, nullptr, nullptr, xW, B_DIM, 3 * D_DIM, D_DIM);
  gru_gemm_wmma<<<dim3(D_DIM / 16, (B_DIM * NS_DIM) / 128), 256, 0, stream>>>(
      S_bf, w_hht, out_alpha, xW, b_ih, b_hh, S, out_S);

  // ---- 5. readout + proj + residual
  readout_kernel<<<(B_DIM * D_DIM) / 256, 256, 0, stream>>>(out_alpha, out_S, sc_bf);
  gemm_wmma_bf16<EP_ADD_F32><<<dim3(D_DIM / 64, B_DIM / 64), 256, 0, stream>>>(
      sc_bf, w_projt, b_proj, x, x_mid, B_DIM, D_DIM, D_DIM);

  // ---- 6. LN1 -> FFN -> +residual -> LN2
  layernorm_kernel<<<B_DIM, 256, 0, stream>>>(x_mid, ln1_g, ln1_b, nullptr, ln1_bf);
  gemm_wmma_bf16<EP_GELU_BF16><<<dim3(H_DIM / 64, B_DIM / 64), 256, 0, stream>>>(
      ln1_bf, w_f1t, b_f1, nullptr, ff1_bf, B_DIM, H_DIM, D_DIM);
  gemm_wmma_bf16<EP_ADD_F32><<<dim3(D_DIM / 64, B_DIM / 64), 256, 0, stream>>>(
      ff1_bf, w_f2t, b_f2, x_mid, tbuf, B_DIM, D_DIM, H_DIM);
  layernorm_kernel<<<B_DIM, 256, 0, stream>>>(tbuf, ln2_g, ln2_b, out_x, nullptr);
}
